// SST_23991687315907
// MI455X (gfx1250) — matrix operand order repaired and checked by reading, not compiled, not run on hardware
//
#include <hip/hip_runtime.h>
#include <hip/hip_bf16.h>

typedef __attribute__((ext_vector_type(16))) __bf16 v16bf;
typedef __attribute__((ext_vector_type(8)))  __bf16 v8bf;
typedef __attribute__((ext_vector_type(8)))  float  v8f;

namespace {
constexpr int kTV = 64, kTC = 20;
constexpr int kHid = 512;
}

__device__ __forceinline__ float sigf(float x) { return 1.0f / (1.0f + expf(-x)); }

__device__ __forceinline__ v16bf cat16(v8bf lo, v8bf hi) {
  return __builtin_shufflevector(lo, hi, 0, 1, 2, 3, 4, 5, 6, 7,
                                 8, 9, 10, 11, 12, 13, 14, 15);
}

// ---------------- f32 -> bf16 pack ----------------
__global__ __launch_bounds__(256) void cvt_bf16_kernel(const float* __restrict__ src,
                                                       __bf16* __restrict__ dst, int n) {
  int i = blockIdx.x * 256 + threadIdx.x;
  if (i < n) dst[i] = (__bf16)src[i];
}

// ---------------- zero-pad inner dim to bf16: (rows, sk) -> (rows, dk) ----------------
__global__ __launch_bounds__(256) void pad_rows_bf16(const float* __restrict__ src,
                                                     __bf16* __restrict__ dst,
                                                     int rows, int sk, int dk) {
  int i = blockIdx.x * 256 + threadIdx.x;
  if (i >= rows * dk) return;
  int r = i / dk, c = i - r * dk;
  dst[i] = (c < sk) ? (__bf16)src[(size_t)r * sk + c] : (__bf16)0.0f;
}

// ---------------- C(M,N) = X(M,K) * W(N,K)^T + bias0 + bias1 ----------------
// X, W pre-packed bf16. 32x64 tile per wave: 2 A-frags x 4 B-frags -> 8 WMMAs
// per 32-K step at 1.5 b128 loads/WMMA; 8 accumulators (dep distance 8).
// Requires M%32==0, N%64==0, K%32==0. Optional f32 and/or bf16 result stores.
__global__ __launch_bounds__(256) void gemm_bf(const __bf16* __restrict__ X,
                                               const __bf16* __restrict__ W,
                                               const float* __restrict__ bias0,
                                               const float* __restrict__ bias1,
                                               float* __restrict__ Cf,
                                               __bf16* __restrict__ Cb,
                                               int M, int N, int K) {
  const int lane  = threadIdx.x & 31;
  const int wave  = threadIdx.x >> 5;
  const int tilesN = N >> 6;
  const int tile  = blockIdx.x * 8 + wave;          // wave-uniform -> full-EXEC WMMA
  if (tile >= (M >> 5) * tilesN) return;
  const int tm   = (tile / tilesN) << 5;
  const int tn   = (tile % tilesN) << 6;
  const int l16  = lane & 15;
  const int half = lane >> 4;

  // A: lane row = tm(+16)+l16; K slice kb+16h..+15
  // B: lane row = tn+16*ni+l16 (W row, contiguous K); slice kb+16h..+15
  const __bf16* __restrict__ xrow0 = X + (size_t)(tm + l16) * K + half * 16;
  const __bf16* __restrict__ xrow1 = X + (size_t)(tm + 16 + l16) * K + half * 16;
  const __bf16* __restrict__ wrow  = W + (size_t)(tn + l16) * K + half * 16;
  const size_t wstep = (size_t)16 * K;

  v8f acc[2][4];
#pragma unroll
  for (int mi = 0; mi < 2; ++mi)
#pragma unroll
    for (int ni = 0; ni < 4; ++ni) acc[mi][ni] = (v8f){};

  for (int kb = 0; kb < K; kb += 32) {
    v16bf a0 = cat16(*(const v8bf*)(xrow0 + kb), *(const v8bf*)(xrow0 + kb + 8));
    v16bf a1 = cat16(*(const v8bf*)(xrow1 + kb), *(const v8bf*)(xrow1 + kb + 8));
    v16bf b[4];
#pragma unroll
    for (int ni = 0; ni < 4; ++ni) b[ni] = *(const v16bf*)(wrow + wstep * ni + kb);
#pragma unroll
    for (int ni = 0; ni < 4; ++ni) {
      acc[0][ni] = __builtin_amdgcn_wmma_f32_16x16x32_bf16(false, a0, false, b[ni],
                                                           (short)0, acc[0][ni], false, false);
      acc[1][ni] = __builtin_amdgcn_wmma_f32_16x16x32_bf16(false, a1, false, b[ni],
                                                           (short)0, acc[1][ni], false, false);
    }
  }

#pragma unroll
  for (int ni = 0; ni < 4; ++ni) {
    float bb = 0.0f;
    if (bias0) bb += bias0[tn + ni * 16 + l16];
    if (bias1) bb += bias1[tn + ni * 16 + l16];
#pragma unroll
    for (int mi = 0; mi < 2; ++mi) {
#pragma unroll
      for (int r = 0; r < 8; ++r) {
        int m = tm + mi * 16 + r + half * 8;        // C/D layout: VGPR r -> M=r (+8 hi lanes)
        size_t idx = (size_t)m * N + tn + ni * 16 + l16;
        float v = acc[mi][ni][r] + bb;
        if (Cf) Cf[idx] = v;
        if (Cb) Cb[idx] = (__bf16)v;
      }
    }
  }
}

// ---------------- persistent single-workgroup LSTM ----------------
// block 0: video (T=64); block 1: caption (T=20). 32 waves; wave w owns hidden
// cols [16w,16w+16) and computes all 4 gate quadrants for them. c in registers,
// h in LDS as bf16 (next step's A operand); f32 h and bf16 h copies to global.
__global__ __launch_bounds__(1024) void lstm_kernel(const float* __restrict__ xg_v,
                                                    const __bf16* __restrict__ whh_v,
                                                    float* __restrict__ hout_v,
                                                    __bf16* __restrict__ houtb_v,
                                                    const float* __restrict__ xg_c,
                                                    const __bf16* __restrict__ whh_c,
                                                    float* __restrict__ hout_c,
                                                    __bf16* __restrict__ houtb_c) {
  const float* xg; const __bf16* whh; float* hout; __bf16* houtb; int T;
  if (blockIdx.x == 0) { xg = xg_v; whh = whh_v; hout = hout_v; houtb = houtb_v; T = kTV; }
  else                 { xg = xg_c; whh = whh_c; hout = hout_c; houtb = houtb_c; T = kTC; }

  __shared__ __bf16 hb[16 * kHid];                  // 16 KB

  const int lane = threadIdx.x & 31;
  const int wave = threadIdx.x >> 5;
  const int l16  = lane & 15;
  const int half = lane >> 4;
  const int j0   = wave * 16;
  const int j    = j0 + l16;

  float creg[8];
#pragma unroll
  for (int r = 0; r < 8; ++r) creg[r] = 0.0f;
  for (int i = threadIdx.x; i < 16 * kHid; i += 1024) hb[i] = (__bf16)0.0f;
  __syncthreads();

  const __bf16* __restrict__ hrow = hb + l16 * kHid + half * 16;
  for (int t = 0; t < T; ++t) {
    v8f acc[4];
#pragma unroll
    for (int q = 0; q < 4; ++q) {                   // C preloaded with xg (bih+bhh folded)
      int n = j0 + q * 512 + l16;
#pragma unroll
      for (int r = 0; r < 8; ++r) {
        int m = r + half * 8;
        acc[q][r] = xg[((size_t)m * T + t) * 2048 + n];
      }
    }
    for (int kb = 0; kb < kHid; kb += 32) {
      v16bf a = cat16(*(const v8bf*)(hrow + kb), *(const v8bf*)(hrow + kb + 8));
#pragma unroll
      for (int q = 0; q < 4; ++q) {                 // 4 independent WMMAs share A
        int n = j0 + q * 512 + l16;
        v16bf b = *(const v16bf*)(whh + (size_t)n * kHid + kb + half * 16);
        acc[q] = __builtin_amdgcn_wmma_f32_16x16x32_bf16(false, a, false, b, (short)0,
                                                         acc[q], false, false);
      }
    }
    __syncthreads();                                // all h reads done before overwrite
#pragma unroll
    for (int r = 0; r < 8; ++r) {
      int m = r + half * 8;
      float iv = acc[0][r], fv = acc[1][r], gv = acc[2][r], ov = acc[3][r];
      float cc = sigf(fv) * creg[r] + sigf(iv) * tanhf(gv);
      float hh = sigf(ov) * tanhf(cc);
      creg[r] = cc;
      hb[m * kHid + j] = (__bf16)hh;
      hout[((size_t)m * T + t) * kHid + j]  = hh;
      houtb[((size_t)m * T + t) * kHid + j] = (__bf16)hh;
    }
    __syncthreads();
  }
}

// ---------------- scores[vt,ck] = w . tanh(vid_lin[vt]+cap_lin[ck]) + b ----------------
__global__ __launch_bounds__(256) void att_scores(const float* __restrict__ vlin,
                                                  const float* __restrict__ clin,
                                                  const float* __restrict__ watt,
                                                  const float* __restrict__ batt,
                                                  float* __restrict__ scores) {
  __shared__ float vl[16][257];
  __shared__ float cl[32][257];
  __shared__ float wa[256];
  const int vt0 = (blockIdx.x / 10) * 16;           // 64 vt tiles
  const int ck0 = (blockIdx.x % 10) * 32;           // 10 ck tiles (320 rows)
  const int tid = threadIdx.x;
  for (int i = tid; i < 16 * 256; i += 256) vl[i >> 8][i & 255] = vlin[(size_t)(vt0 + (i >> 8)) * 256 + (i & 255)];
  for (int i = tid; i < 32 * 256; i += 256) cl[i >> 8][i & 255] = clin[(size_t)(ck0 + (i >> 8)) * 256 + (i & 255)];
  wa[tid] = watt[tid];
  __syncthreads();
  const float b0 = batt[0];
#pragma unroll
  for (int s = 0; s < 2; ++s) {
    int sid = tid + 256 * s;
    int vti = sid >> 5, cki = sid & 31;
    float acc = 0.0f;
    for (int a = 0; a < 256; ++a) acc += wa[a] * tanhf(vl[vti][a] + cl[cki][a]);
    scores[(size_t)(vt0 + vti) * 320 + ck0 + cki] = acc + b0;
  }
}

// ---------------- masked softmax(k) + atten + cosine sim; one wave per (b,t,c) ----------------
__global__ __launch_bounds__(256) void att_apply(const float* __restrict__ scores,
                                                 const float* __restrict__ caph,
                                                 const float* __restrict__ vidh,
                                                 const int* __restrict__ lens,
                                                 float* __restrict__ simbuf) {
  __shared__ float wsm[8][32];
  const int lane = threadIdx.x & 31;
  const int lw   = threadIdx.x >> 5;
  const int wid  = blockIdx.x * 8 + lw;             // 0..16383
  const int vt   = wid >> 4;
  const int c    = wid & 15;

  int L = lens[c]; if (L < 1) L = 1; if (L > kTC) L = kTC;
  float sc = -1e30f;
  if (lane < kTC) {
    sc = scores[(size_t)vt * 320 + c * kTC + lane];
    if (lane >= L) sc = -1e30f;
  }
  float m = sc;
  for (int o = 16; o > 0; o >>= 1) m = fmaxf(m, __shfl_xor(m, o, 32));
  float e = (lane < kTC) ? expf(sc - m) : 0.0f;
  float s = e;
  for (int o = 16; o > 0; o >>= 1) s += __shfl_xor(s, o, 32);
  wsm[lw][lane] = e / s;
  __syncthreads();

  float num = 0.0f, na = 0.0f, nv = 0.0f;
  const float* __restrict__ ch = caph + (size_t)c * kTC * kHid;
  const float* __restrict__ vh = vidh + (size_t)vt * kHid;
  for (int h = lane; h < kHid; h += 32) {
    float a = 0.0f;
#pragma unroll
    for (int k = 0; k < kTC; ++k) a += wsm[lw][k] * ch[k * kHid + h];
    float v = vh[h];
    num += a * v; na += a * a; nv += v * v;
  }
  for (int o = 16; o > 0; o >>= 1) {
    num += __shfl_xor(num, o, 32);
    na  += __shfl_xor(na,  o, 32);
    nv  += __shfl_xor(nv,  o, 32);
  }
  if (lane == 0) {
    float den = fmaxf(sqrtf(na), 1e-6f) * fmaxf(sqrtf(nv), 1e-6f);
    simbuf[wid] = num / den;
  }
}

// ---------------- out[b,c] = mean_t sim[b,t,c] ----------------
__global__ __launch_bounds__(256) void mean_t_kernel(const float* __restrict__ simbuf,
                                                     float* __restrict__ out) {
  int i = blockIdx.x * 256 + threadIdx.x;
  if (i < 256) {
    int b = i >> 4, c = i & 15;
    float s = 0.0f;
    for (int t = 0; t < kTV; ++t) s += simbuf[((b * kTV + t) * 16 + c)];
    out[i] = s * (1.0f / 64.0f);
  }
}

extern "C" void kernel_launch(void* const* d_in, const int* in_sizes, int n_in,
                              void* d_out, int out_size, void* d_ws, size_t ws_size,
                              hipStream_t stream) {
  const float* video_fc = (const float*)d_in[0];    // (16,64,2048)
  const float* caption  = (const float*)d_in[1];    // (16,20,300)
  const int*   lens     = (const int*)  d_in[2];    // (16,)
  const float* W_ve  = (const float*)d_in[3];  const float* b_ve  = (const float*)d_in[4];
  const float* W_we  = (const float*)d_in[5];  const float* b_we  = (const float*)d_in[6];
  const float* Wih_v = (const float*)d_in[7];  const float* Whh_v = (const float*)d_in[8];
  const float* bih_v = (const float*)d_in[9];  const float* bhh_v = (const float*)d_in[10];
  const float* Wih_c = (const float*)d_in[11]; const float* Whh_c = (const float*)d_in[12];
  const float* bih_c = (const float*)d_in[13]; const float* bhh_c = (const float*)d_in[14];
  const float* W_vl  = (const float*)d_in[15]; const float* b_vl  = (const float*)d_in[16];
  const float* W_sl  = (const float*)d_in[17]; const float* b_sl  = (const float*)d_in[18];
  const float* w_att = (const float*)d_in[19]; const float* b_att = (const float*)d_in[20];
  float* out = (float*)d_out;

  // workspace layout (float slots; bf16 buffers use elems/2 slots) — ~35 MB, L2-resident
  float* ws = (float*)d_ws;
  float*  vid_xg   = ws + 0;                    // 1024x2048 f32
  float*  cap_xg   = ws + 2097152;              // 320x2048  f32
  float*  vid_h    = ws + 2752512;              // 1024x512  f32
  float*  cap_h    = ws + 3276800;              // 320x512   f32
  float*  vid_lin  = ws + 3440640;              // 1024x256  f32
  float*  cap_lin  = ws + 3702784;              // 320x256   f32
  float*  scores   = ws + 3784704;              // 1024x320  f32
  float*  simbuf   = ws + 4112384;              // 16384     f32
  __bf16* whhv_bf  = (__bf16*)(ws + 4128768);   // 2048x512
  __bf16* whhc_bf  = (__bf16*)(ws + 4653056);   // 2048x512
  __bf16* wve_bf   = (__bf16*)(ws + 5177344);   // 512x2048
  __bf16* wihv_bf  = (__bf16*)(ws + 5701632);   // 2048x512
  __bf16* wihc_bf  = (__bf16*)(ws + 6225920);   // 2048x512
  __bf16* wvl_bf   = (__bf16*)(ws + 6750208);   // 256x512
  __bf16* wsl_bf   = (__bf16*)(ws + 6815744);   // 256x512
  __bf16* vfc_bf   = (__bf16*)(ws + 6881280);   // 1024x2048
  __bf16* cappad_bf= (__bf16*)(ws + 7929856);   // 320x320 (K padded 300->320)
  __bf16* wwepad_bf= (__bf16*)(ws + 7981056);   // 512x320
  __bf16* vidx_bf  = (__bf16*)(ws + 8062976);   // 1024x512
  __bf16* capx_bf  = (__bf16*)(ws + 8325120);   // 320x512
  __bf16* vidh_bf  = (__bf16*)(ws + 8407040);   // 1024x512
  __bf16* caph_bf  = (__bf16*)(ws + 8669184);   // 320x512
  // end: 8751104 float slots

  // pre-pack every GEMM operand to bf16 (one pass; all later reads are 2B/elem, 0 cvt)
  cvt_bf16_kernel<<<4096, 256, 0, stream>>>(Whh_v,    whhv_bf, 2048 * 512);
  cvt_bf16_kernel<<<4096, 256, 0, stream>>>(Whh_c,    whhc_bf, 2048 * 512);
  cvt_bf16_kernel<<<4096, 256, 0, stream>>>(W_ve,     wve_bf,  512 * 2048);
  cvt_bf16_kernel<<<4096, 256, 0, stream>>>(Wih_v,    wihv_bf, 2048 * 512);
  cvt_bf16_kernel<<<4096, 256, 0, stream>>>(Wih_c,    wihc_bf, 2048 * 512);
  cvt_bf16_kernel<<<512,  256, 0, stream>>>(W_vl,     wvl_bf,  256 * 512);
  cvt_bf16_kernel<<<512,  256, 0, stream>>>(W_sl,     wsl_bf,  256 * 512);
  cvt_bf16_kernel<<<8192, 256, 0, stream>>>(video_fc, vfc_bf,  1024 * 2048);
  pad_rows_bf16<<<400, 256, 0, stream>>>(caption, cappad_bf, 320, 300, 320);
  pad_rows_bf16<<<640, 256, 0, stream>>>(W_we,    wwepad_bf, 512, 300, 320);

  // embedding GEMMs (bf16 out only: they feed the next GEMM)
  gemm_bf<<<32, 256, 0, stream>>>(vfc_bf,    wve_bf,    b_ve, nullptr, nullptr, vidx_bf, 1024, 512, 2048);
  gemm_bf<<<10, 256, 0, stream>>>(cappad_bf, wwepad_bf, b_we, nullptr, nullptr, capx_bf, 320,  512, 320);
  // gate-input GEMMs with both LSTM biases folded in (f32 out: LSTM C preload)
  gemm_bf<<<128, 256, 0, stream>>>(vidx_bf, wihv_bf, bih_v, bhh_v, vid_xg, nullptr, 1024, 2048, 512);
  gemm_bf<<<40,  256, 0, stream>>>(capx_bf, wihc_bf, bih_c, bhh_c, cap_xg, nullptr, 320,  2048, 512);

  // both LSTMs concurrently: block 0 = video (64 steps), block 1 = caption (20 steps)
  lstm_kernel<<<2, 1024, 0, stream>>>(vid_xg, whhv_bf, vid_h, vidh_bf,
                                      cap_xg, whhc_bf, cap_h, caph_bf);

  // attention projections (f32 out: consumed by tanh/score kernels)
  gemm_bf<<<16, 256, 0, stream>>>(vidh_bf, wvl_bf, b_vl, nullptr, vid_lin, nullptr, 1024, 256, 512);
  gemm_bf<<<5,  256, 0, stream>>>(caph_bf, wsl_bf, b_sl, nullptr, cap_lin, nullptr, 320,  256, 512);

  // scores, masked softmax + cosine similarity, temporal mean
  att_scores<<<640,  256, 0, stream>>>(vid_lin, cap_lin, w_att, b_att, scores);
  att_apply <<<2048, 256, 0, stream>>>(scores, cap_h, vid_h, lens, simbuf);
  mean_t_kernel<<<1, 256, 0, stream>>>(simbuf, out);
}
